// MultiHeadSelfAttention_21414706938578
// MI455X (gfx1250) — matrix operand order repaired and checked by reading, not compiled, not run on hardware
//
#include <hip/hip_runtime.h>
#include <hip/hip_bf16.h>

typedef __attribute__((ext_vector_type(16))) _Float16 v16h;
typedef __attribute__((ext_vector_type(8)))  _Float16 v8h;
typedef __attribute__((ext_vector_type(8)))  float    v8f;
typedef __attribute__((ext_vector_type(4)))  float    v4f;

#define Bsz 4
#define Cseq 2048
#define Emb 1024
#define Hn 16
#define Dh 64
#define NEGV -1e9f

static __device__ __forceinline__ v16h cat8(v8h a, v8h b) {
  return __builtin_shufflevector(a, b, 0,1,2,3,4,5,6,7,8,9,10,11,12,13,14,15);
}
// contiguous 16 halfs (B operand row chunk; lane = col n, K half picked by lane<16)
static __device__ __forceinline__ v16h ld16h(const _Float16* p) {
  v8h a = *(const v8h*)p;
  v8h b = *(const v8h*)(p + 8);
  return cat8(a, b);
}
// A operand: 16 contiguous halfs at +2*off0 (contiguous K order, matches B)
static __device__ __forceinline__ v16h ldA(const _Float16* row, int k, int off0) {
  v8h a = *(const v8h*)(row + k + 2 * off0);
  v8h b = *(const v8h*)(row + k + 2 * off0 + 8);
  return cat8(a, b);
}
static __device__ __forceinline__ v8f wmma16(v16h a, v16h b, v8f c) {
  return __builtin_amdgcn_wmma_f32_16x16x32_f16(false, a, false, b, (short)0, c, false, false);
}

// ---------------- stage 0: conversions ----------------
// 8 contiguous f32 -> 8 f16 per thread (b128 in, b128 out)
__global__ void cvt_f16_kernel(const float* __restrict__ src, _Float16* __restrict__ dst, int n8) {
  int i = blockIdx.x * 256 + threadIdx.x;
  if (i >= n8) return;
  v4f a = *(const v4f*)(src + i * 8);
  v4f b = *(const v4f*)(src + i * 8 + 4);
  v8h o;
#pragma unroll
  for (int e = 0; e < 4; ++e) { o[e] = (_Float16)a[e]; o[e + 4] = (_Float16)b[e]; }
  *(v8h*)(dst + i * 8) = o;
}
// src[R][Cc] f32 -> dst[Cc][R] f16 (dst writes coalesced)
__global__ void transpose_cvt_kernel(const float* __restrict__ src, _Float16* __restrict__ dst,
                                     int R, int Cc) {
  int i = blockIdx.x * 256 + threadIdx.x;
  if (i >= R * Cc) return;
  int n = i / R, k = i % R;
  dst[i] = (_Float16)src[k * Cc + n];
}

// ---------------- stage 1: QKV GEMM ----------------
// xh [8192][1024] f16, WT [3072][1024] f16 (= W_qkv^T).
// One wave -> 16M x 64N tile; explicit 2-stage software pipeline (no reg rotation).
__global__ __launch_bounds__(256, 1) void qkv_gemm_kernel(
    const _Float16* __restrict__ xh,
    const _Float16* __restrict__ WT,
    const float* __restrict__ bqkv,
    _Float16* __restrict__ Qb,
    _Float16* __restrict__ Kb,
    _Float16* __restrict__ VTb) {
  const int lane = threadIdx.x & 31;
  const int wave = threadIdx.x >> 5;
  const int job  = blockIdx.x * 8 + wave;          // 512 * 48 jobs
  const int mt = job / 48, nt = job % 48;
  const int mbase = mt * 16, nbase = nt * 64;
  const int ln   = lane & 15;
  const int aoff0  = (lane < 16) ? 0 : 8;
  const int boff   = (lane < 16) ? 0 : 16;
  const int rowoff = (lane < 16) ? 0 : 8;

  const _Float16* arow = xh + (mbase + ln) * Emb;
  const _Float16* brow = WT + (nbase + ln) * Emb + boff;   // + t*16*Emb per n-tile

  v8f acc[4] = {v8f{}, v8f{}, v8f{}, v8f{}};

  v16h Aa = ldA(arow, 0, aoff0);
  v16h Ba[4];
#pragma unroll
  for (int t = 0; t < 4; ++t) Ba[t] = ld16h(brow + t * 16 * Emb);

  for (int kk = 0; kk < Emb - 64; kk += 64) {
    v16h Ab = ldA(arow, kk + 32, aoff0);
    v16h Bb[4];
#pragma unroll
    for (int t = 0; t < 4; ++t) Bb[t] = ld16h(brow + t * 16 * Emb + kk + 32);
#pragma unroll
    for (int t = 0; t < 4; ++t) acc[t] = wmma16(Aa, Ba[t], acc[t]);

    Aa = ldA(arow, kk + 64, aoff0);
#pragma unroll
    for (int t = 0; t < 4; ++t) Ba[t] = ld16h(brow + t * 16 * Emb + kk + 64);
#pragma unroll
    for (int t = 0; t < 4; ++t) acc[t] = wmma16(Ab, Bb[t], acc[t]);
  }
  {   // tail: k-steps Emb-64 (stage a, preloaded) and Emb-32 (stage b)
    v16h Ab = ldA(arow, Emb - 32, aoff0);
    v16h Bb[4];
#pragma unroll
    for (int t = 0; t < 4; ++t) Bb[t] = ld16h(brow + t * 16 * Emb + Emb - 32);
#pragma unroll
    for (int t = 0; t < 4; ++t) acc[t] = wmma16(Aa, Ba[t], acc[t]);
#pragma unroll
    for (int t = 0; t < 4; ++t) acc[t] = wmma16(Ab, Bb[t], acc[t]);
  }

#pragma unroll
  for (int t = 0; t < 4; ++t) {
    int n = nbase + t * 16 + ln;
    float bias = bqkv[n];
    int which = n >> 10;           // 0=Q 1=K 2=V
    int nn = n & 1023;
    int h = nn >> 6, d = nn & 63;
#pragma unroll
    for (int r = 0; r < 8; ++r) {
      int mrow = mbase + r + rowoff;
      int b = mrow >> 11, c = mrow & 2047;
      float v = acc[t][r] + bias;
      int idx = ((b * Hn + h) * Cseq + c) * Dh + d;
      if (which == 0)      Qb[idx] = (_Float16)(v * 0.125f);   // fold 1/sqrt(64)
      else if (which == 1) Kb[idx] = (_Float16)v;
      else VTb[((b * Hn + h) * Dh + d) * Cseq + c] = (_Float16)v;
    }
  }
}

// ---------------- stage 2: flash attention ----------------
// One wave per (b,h, 16 query rows). 32 keys per iteration.
__global__ __launch_bounds__(256, 1) void attn_kernel(
    const _Float16* __restrict__ Qb,
    const _Float16* __restrict__ Kb,
    const _Float16* __restrict__ VTb,
    const unsigned char* __restrict__ pmask,
    _Float16* __restrict__ AO) {
  __shared__ __align__(16) _Float16 Pl[8][16 * 32];
  const int lane = threadIdx.x & 31;
  const int wave = threadIdx.x >> 5;
  const int job  = blockIdx.x * 8 + wave;          // 64 * 128 jobs
  const int bh = job >> 7;
  const int qb = (job & 127) << 4;
  const int b  = bh >> 4, h = bh & 15;
  const int ln   = lane & 15;
  const int aoff0  = (lane < 16) ? 0 : 8;
  const int koff   = (lane < 16) ? 0 : 16;
  const int rowoff = (lane < 16) ? 0 : 8;

  const _Float16* qrow = Qb + (bh * Cseq + qb + ln) * Dh;
  v16h A0 = ldA(qrow, 0, aoff0);                   // d 0..31
  v16h A1 = ldA(qrow, 32, aoff0);                  // d 32..63

  float m[8], l[8];
#pragma unroll
  for (int r = 0; r < 8; ++r) { m[r] = -3.0e38f; l[r] = 0.0f; }
  v8f O[4] = {v8f{}, v8f{}, v8f{}, v8f{}};

  const unsigned char* pmb = pmask + b * Cseq;
  const _Float16* krow = Kb + (bh * Cseq + ln) * Dh + koff;
  const _Float16* vrow = VTb + (bh * Dh + ln) * Cseq + koff;
  _Float16* myP = &Pl[wave][0];

  for (int jb = 0; jb < Cseq; jb += 32) {
    const bool v0 = pmb[jb + ln] != 0, v1 = pmb[jb + 16 + ln] != 0;
    if (__ballot(v0 || v1) == 0ULL) continue;      // whole 32-key block padded

    // K tiles (S depends on these)
    const _Float16* kr0 = krow + jb * Dh;
    const _Float16* kr1 = kr0 + 16 * Dh;
    v16h BK00 = ld16h(kr0);
    v16h BK01 = ld16h(kr0 + 32);
    v16h BK10 = ld16h(kr1);
    v16h BK11 = ld16h(kr1 + 32);
    // V tiles issued early: latency hides behind S WMMAs + softmax below
    v16h BV[4];
#pragma unroll
    for (int t = 0; t < 4; ++t) BV[t] = ld16h(vrow + t * 16 * Cseq + jb);

    v8f S0 = {}, S1 = {};
    S0 = wmma16(A0, BK00, S0);
    S0 = wmma16(A1, BK01, S0);
    S1 = wmma16(A0, BK10, S1);
    S1 = wmma16(A1, BK11, S1);

#pragma unroll
    for (int r = 0; r < 8; ++r) {
      float x0 = v0 ? S0[r] : NEGV;
      float x1 = v1 ? S1[r] : NEGV;
      float mx = fmaxf(x0, x1);
      mx = fmaxf(mx, __shfl_xor(mx, 1));
      mx = fmaxf(mx, __shfl_xor(mx, 2));
      mx = fmaxf(mx, __shfl_xor(mx, 4));
      mx = fmaxf(mx, __shfl_xor(mx, 8));
      float mn = fmaxf(m[r], mx);
      float corr = __expf(m[r] - mn);
      m[r] = mn;
      float p0 = __expf(x0 - mn);
      float p1 = __expf(x1 - mn);
      float rs = p0 + p1;
      rs += __shfl_xor(rs, 1);
      rs += __shfl_xor(rs, 2);
      rs += __shfl_xor(rs, 4);
      rs += __shfl_xor(rs, 8);
      l[r] = l[r] * corr + rs;
      O[0][r] *= corr; O[1][r] *= corr; O[2][r] *= corr; O[3][r] *= corr;
      myP[(r + rowoff) * 32 + ln]      = (_Float16)p0;
      myP[(r + rowoff) * 32 + 16 + ln] = (_Float16)p1;
    }
    asm volatile("s_wait_dscnt 0x0" ::: "memory");
    v16h PA = ldA(myP + ln * 32, 0, aoff0);        // P tile in A-operand layout

#pragma unroll
    for (int t = 0; t < 4; ++t) O[t] = wmma16(PA, BV[t], O[t]);
    asm volatile("" ::: "memory");                 // keep next stores after reads
  }

#pragma unroll
  for (int r = 0; r < 8; ++r) {
    float inv = 1.0f / l[r];
    int q = qb + r + rowoff;
#pragma unroll
    for (int t = 0; t < 4; ++t)
      AO[(b * Cseq + q) * Emb + h * Dh + t * 16 + ln] = (_Float16)(O[t][r] * inv);
  }
}

// ---------------- stage 3: output projection ----------------
// One wave -> 16M x 64N tile; 2-stage pipeline; f32 out + bias.
__global__ __launch_bounds__(256, 1) void out_gemm_kernel(
    const _Float16* __restrict__ AO,
    const _Float16* __restrict__ WT,   // W_out^T [1024][1024]
    const float* __restrict__ bout,
    float* __restrict__ out) {
  const int lane = threadIdx.x & 31;
  const int wave = threadIdx.x >> 5;
  const int job  = blockIdx.x * 8 + wave;          // 512 * 16 jobs
  const int mt = job / 16, nt = job % 16;
  const int mbase = mt * 16, nbase = nt * 64;
  const int ln   = lane & 15;
  const int aoff0  = (lane < 16) ? 0 : 8;
  const int boff   = (lane < 16) ? 0 : 16;
  const int rowoff = (lane < 16) ? 0 : 8;

  const _Float16* arow = AO + (mbase + ln) * Emb;
  const _Float16* brow = WT + (nbase + ln) * Emb + boff;

  v8f acc[4] = {v8f{}, v8f{}, v8f{}, v8f{}};

  v16h Aa = ldA(arow, 0, aoff0);
  v16h Ba[4];
#pragma unroll
  for (int t = 0; t < 4; ++t) Ba[t] = ld16h(brow + t * 16 * Emb);

  for (int kk = 0; kk < Emb - 64; kk += 64) {
    v16h Ab = ldA(arow, kk + 32, aoff0);
    v16h Bb[4];
#pragma unroll
    for (int t = 0; t < 4; ++t) Bb[t] = ld16h(brow + t * 16 * Emb + kk + 32);
#pragma unroll
    for (int t = 0; t < 4; ++t) acc[t] = wmma16(Aa, Ba[t], acc[t]);

    Aa = ldA(arow, kk + 64, aoff0);
#pragma unroll
    for (int t = 0; t < 4; ++t) Ba[t] = ld16h(brow + t * 16 * Emb + kk + 64);
#pragma unroll
    for (int t = 0; t < 4; ++t) acc[t] = wmma16(Ab, Bb[t], acc[t]);
  }
  {
    v16h Ab = ldA(arow, Emb - 32, aoff0);
    v16h Bb[4];
#pragma unroll
    for (int t = 0; t < 4; ++t) Bb[t] = ld16h(brow + t * 16 * Emb + Emb - 32);
#pragma unroll
    for (int t = 0; t < 4; ++t) acc[t] = wmma16(Aa, Ba[t], acc[t]);
#pragma unroll
    for (int t = 0; t < 4; ++t) acc[t] = wmma16(Ab, Bb[t], acc[t]);
  }

#pragma unroll
  for (int t = 0; t < 4; ++t) {
    int n = nbase + t * 16 + ln;
    float bias = bout[n];
#pragma unroll
    for (int r = 0; r < 8; ++r)
      out[(mbase + r + rowoff) * Emb + n] = acc[t][r] + bias;
  }
}

extern "C" void kernel_launch(void* const* d_in, const int* in_sizes, int n_in,
                              void* d_out, int out_size, void* d_ws, size_t ws_size,
                              hipStream_t stream) {
  const float* x  = (const float*)d_in[0];
  const unsigned char* pm = (const unsigned char*)d_in[1];   // bool mask [B,C]
  const float* Wqkv = (const float*)d_in[2];
  const float* bqkv = (const float*)d_in[3];
  const float* Wout = (const float*)d_in[4];
  const float* bout = (const float*)d_in[5];
  float* out = (float*)d_out;

  char* w = (char*)d_ws;
  _Float16* xh    = (_Float16*)w;  w += (size_t)Bsz * Cseq * Emb * 2;        // 16 MB
  _Float16* WqkvT = (_Float16*)w;  w += (size_t)3 * Emb * Emb * 2;           // 6 MB
  _Float16* WoutT = (_Float16*)w;  w += (size_t)Emb * Emb * 2;               // 2 MB
  _Float16* Qb    = (_Float16*)w;  w += (size_t)Bsz * Hn * Cseq * Dh * 2;    // 16 MB
  _Float16* Kb    = (_Float16*)w;  w += (size_t)Bsz * Hn * Cseq * Dh * 2;    // 16 MB
  _Float16* VTb   = (_Float16*)w;  w += (size_t)Bsz * Hn * Dh * Cseq * 2;    // 16 MB
  _Float16* AO    = (_Float16*)w;  w += (size_t)Bsz * Cseq * Emb * 2;        // 16 MB

  const int nx8 = (Bsz * Cseq * Emb) / 8;               // 1048576
  cvt_f16_kernel<<<(nx8 + 255) / 256, 256, 0, stream>>>(x, xh, nx8);
  transpose_cvt_kernel<<<(Emb * 3 * Emb + 255) / 256, 256, 0, stream>>>(Wqkv, WqkvT, Emb, 3 * Emb);
  transpose_cvt_kernel<<<(Emb * Emb + 255) / 256, 256, 0, stream>>>(Wout, WoutT, Emb, Emb);

  qkv_gemm_kernel<<<(512 * 48) / 8, 256, 0, stream>>>(xh, WqkvT, bqkv, Qb, Kb, VTb);
  attn_kernel<<<(64 * 128) / 8, 256, 0, stream>>>(Qb, Kb, VTb, pm, AO);
  out_gemm_kernel<<<(512 * 16) / 8, 256, 0, stream>>>(AO, WoutT, bout, out);
}
